// Decoder_52682068853476
// MI455X (gfx1250) — compile-verified, matrix operand order checked by reading, not run on hardware
//
#include <hip/hip_runtime.h>
#include <hip/hip_bf16.h>

// ---------------------------------------------------------------------------
// Attention-LSTM decoder for MI455X (gfx1250, wave32, WMMA).
// Strategy: convert all weights to bf16 once (L2-resident across the 64
// sequential steps; Wout_bf16 = 64MB << 192MB L2), run every GEMM through
// v_wmma_f32_16x16x32_bf16 with fp32 accumulate, keep softmax / LSTM cell
// in fp32 VALU.  Each wave computes a 32x32 output tile (2 M-tiles x
// 2 N-tiles) so the A fragments are reused across two B fragments:
// 8 loads -> 4 WMMAs per 32-wide K step.
// ---------------------------------------------------------------------------

#define B_  32
#define TDEC 64
#define TENC 128
#define V_  32000
#define E_  512
#define H_  1024
#define A_  128
#define RNN_IN 1664            // E + A + H
#define XH_W   2688            // RNN_IN + H  (concat [x | h])
#define G4H    4096            // 4*H

typedef __attribute__((ext_vector_type(16))) __bf16 v16bf;
typedef __attribute__((ext_vector_type(8)))  float  v8f;

union FragU { uint4 u4[2]; v16bf v; };

__device__ __forceinline__ unsigned short f32_to_bf16(float f) {
  unsigned u = __float_as_uint(f);
  unsigned r = u + 0x7FFFu + ((u >> 16) & 1u);   // round-to-nearest-even
  return (unsigned short)(r >> 16);
}

// Load one 16x32 bf16 fragment (A layout) or a 32x16 B fragment whose
// columns are contiguous rows of a row-major (N x K) weight matrix.
// lanes 0-15:  rows rbase+lane, K = kb+0..7  and kb+16..23
// lanes 16-31: rows rbase+lane, K = kb+8..15 and kb+24..31
__device__ __forceinline__ v16bf load_frag(const unsigned short* __restrict__ base,
                                           int row_stride, int rbase, int kb, int lane) {
  int r    = rbase + (lane & 15);
  int koff = kb + ((lane & 16) ? 8 : 0);
  const unsigned short* p = base + (long)r * row_stride + koff;
  FragU f;
  f.u4[0] = *(const uint4*)(p);
  f.u4[1] = *(const uint4*)(p + 16);
  return f.v;
}

// ---------------------------------------------------------------------------
// Generic skinny GEMM:  out[m][n] = sum_k A[m][k] * W[n][k]  (+ bias[n])
// A: 32 x K bf16 (row stride lda), W: N x K bf16 row-major, out fp32 with
// row stride out_mstride.  256 threads = 8 wave32; each wave owns 32 N-cols
// (two 16-col tiles) and the full M=32 (four accumulators).  K % 32 == 0,
// N % 256 == 0.
// ---------------------------------------------------------------------------
__global__ __launch_bounds__(256) void k_gemm_bf16(
    const unsigned short* __restrict__ A, int lda,
    const unsigned short* __restrict__ W, int K,
    const float* __restrict__ bias,
    float* __restrict__ out, long out_mstride, int N)
{
  int lane = threadIdx.x & 31;
  int wave = threadIdx.x >> 5;
  int nt0  = blockIdx.x * 256 + wave * 32;   // tiles nt0 and nt0+16
  if (nt0 >= N) return;

  v8f c00 = {}; v8f c01 = {}; v8f c10 = {}; v8f c11 = {};
  for (int kb = 0; kb < K; kb += 32) {
    v16bf a0 = load_frag(A, lda, 0,        kb, lane);
    v16bf a1 = load_frag(A, lda, 16,       kb, lane);
    v16bf b0 = load_frag(W, K,   nt0,      kb, lane);
    v16bf b1 = load_frag(W, K,   nt0 + 16, kb, lane);
    c00 = __builtin_amdgcn_wmma_f32_16x16x32_bf16(false, a0, false, b0, (short)0, c00, false, false);
    c10 = __builtin_amdgcn_wmma_f32_16x16x32_bf16(false, a1, false, b0, (short)0, c10, false, false);
    c01 = __builtin_amdgcn_wmma_f32_16x16x32_bf16(false, a0, false, b1, (short)0, c01, false, false);
    c11 = __builtin_amdgcn_wmma_f32_16x16x32_bf16(false, a1, false, b1, (short)0, c11, false, false);
  }

  int col0 = nt0 + (lane & 15);
  int col1 = col0 + 16;
  float bv0 = bias ? bias[col0] : 0.0f;
  float bv1 = bias ? bias[col1] : 0.0f;
  int mlo = (lane & 16) ? 8 : 0;
#pragma unroll
  for (int r = 0; r < 8; ++r) {
    long m0 = (long)(mlo + r) * out_mstride;
    long m1 = (long)(mlo + r + 16) * out_mstride;
    out[m0 + col0] = c00[r] + bv0;
    out[m1 + col0] = c10[r] + bv0;
    out[m0 + col1] = c01[r] + bv1;
    out[m1 + col1] = c11[r] + bv1;
  }
}

// ---------------------------------------------------------------------------
// Attention (one block per batch row, 128 threads = one per encoder pos):
// scores = q . enc + mask_bias ; softmax ; ctx = attn . enc (fp32).
// Also copies this step's embedding (bf16) into xh[:,0:512] and writes
// ctx as bf16 into xh[:,640:1664].
// ---------------------------------------------------------------------------
__global__ __launch_bounds__(128) void k_attn(
    const float* __restrict__ q,          // (32,1024)
    const float* __restrict__ enc,        // (32,128,1024)
    const float* __restrict__ mask_bias,  // (32,128)
    const unsigned short* __restrict__ embs, // (32,64,512) bf16
    unsigned short* __restrict__ xh,      // (32,2688) bf16
    int t)
{
  int b  = blockIdx.x;
  int tp = threadIdx.x;
  __shared__ float sc[TENC];
  __shared__ float red[TENC];

  // stage embedding for this step
  const unsigned short* eb = embs + ((long)b * TDEC + t) * E_;
  unsigned short* xb = xh + (long)b * XH_W;
  for (int j = tp; j < E_; j += TENC) xb[j] = eb[j];

  // score for encoder position tp
  const float* qb = q + (long)b * H_;
  const float* er = enc + ((long)b * TENC + tp) * H_;
  float s = 0.f;
  for (int i = 0; i < H_; i += 4) {
    float4 qv = *(const float4*)(qb + i);
    float4 ev = *(const float4*)(er + i);
    s += qv.x * ev.x + qv.y * ev.y + qv.z * ev.z + qv.w * ev.w;
  }
  s += mask_bias[b * TENC + tp];

  // softmax over 128 positions
  red[tp] = s;
  __syncthreads();
  for (int off = 64; off > 0; off >>= 1) {
    if (tp < off) red[tp] = fmaxf(red[tp], red[tp + off]);
    __syncthreads();
  }
  float m = red[0];
  __syncthreads();
  float p = __expf(s - m);
  red[tp] = p;
  __syncthreads();
  for (int off = 64; off > 0; off >>= 1) {
    if (tp < off) red[tp] += red[tp + off];
    __syncthreads();
  }
  float inv = 1.0f / red[0];
  __syncthreads();
  sc[tp] = p * inv;
  __syncthreads();

  // ctx: each thread owns 8 h-columns (tp, tp+128, ...)
  float acc[8] = {0.f, 0.f, 0.f, 0.f, 0.f, 0.f, 0.f, 0.f};
  for (int tp2 = 0; tp2 < TENC; ++tp2) {
    float a = sc[tp2];
    const float* e2 = enc + ((long)b * TENC + tp2) * H_;
#pragma unroll
    for (int j = 0; j < 8; ++j) acc[j] += a * e2[tp + j * TENC];
  }
#pragma unroll
  for (int j = 0; j < 8; ++j)
    xb[E_ + A_ + tp + j * TENC] = f32_to_bf16(acc[j]);
}

// ---------------------------------------------------------------------------
// LSTM cell: gates -> (h2, c2).  h2 written as bf16 into xh[:,1664:2688]
// (which is the A matrix of the next q-GEMM and of the logits GEMM).
// ---------------------------------------------------------------------------
__global__ __launch_bounds__(256) void k_cell(
    const float* __restrict__ gates,  // (32,4096)
    const float* __restrict__ bih, const float* __restrict__ bhh,
    float* __restrict__ c,            // (32,1024)
    unsigned short* __restrict__ xh)  // (32,2688)
{
  int idx = blockIdx.x * 256 + threadIdx.x;
  if (idx >= B_ * H_) return;
  int b = idx >> 10, j = idx & 1023;
  const float* g = gates + (long)b * G4H;
  float iv = g[j]            + bih[j]            + bhh[j];
  float fv = g[H_ + j]       + bih[H_ + j]       + bhh[H_ + j];
  float gv = g[2 * H_ + j]   + bih[2 * H_ + j]   + bhh[2 * H_ + j];
  float ov = g[3 * H_ + j]   + bih[3 * H_ + j]   + bhh[3 * H_ + j];
  float si = 1.f / (1.f + __expf(-iv));
  float sf = 1.f / (1.f + __expf(-fv));
  float so = 1.f / (1.f + __expf(-ov));
  float c2 = sf * c[idx] + si * tanhf(gv);
  float h2 = so * tanhf(c2);
  c[idx] = c2;
  xh[(long)b * XH_W + RNN_IN + j] = f32_to_bf16(h2);
}

// ---------------------------------------------------------------------------
// One-time (per call) preprocessing kernels
// ---------------------------------------------------------------------------
__global__ void k_cvt(const float* __restrict__ src, unsigned short* __restrict__ dst, long n) {
  long i = (long)blockIdx.x * blockDim.x + threadIdx.x;
  long stride = (long)gridDim.x * blockDim.x;
  for (; i < n; i += stride) dst[i] = f32_to_bf16(src[i]);
}

__global__ void k_pack_wg(const float* __restrict__ Wih, const float* __restrict__ Whh,
                          unsigned short* __restrict__ Wg) {
  long n = (long)G4H * XH_W;
  long i = (long)blockIdx.x * blockDim.x + threadIdx.x;
  long stride = (long)gridDim.x * blockDim.x;
  for (; i < n; i += stride) {
    long r = i / XH_W, cc = i % XH_W;
    float v = (cc < RNN_IN) ? Wih[r * RNN_IN + cc] : Whh[r * H_ + (cc - RNN_IN)];
    Wg[i] = f32_to_bf16(v);
  }
}

__global__ void k_transpose_wa(const float* __restrict__ Wa, unsigned short* __restrict__ WaT) {
  long n = (long)H_ * H_;
  long i = (long)blockIdx.x * blockDim.x + threadIdx.x;
  long stride = (long)gridDim.x * blockDim.x;
  for (; i < n; i += stride) {
    long nn = i / H_, kk = i % H_;          // WaT[nn][kk] = Wa[kk][nn]
    WaT[i] = f32_to_bf16(Wa[kk * H_ + nn]);
  }
}

__global__ void k_init(const int* __restrict__ input_var, const float* __restrict__ embed,
                       const float* __restrict__ add_var, const float* __restrict__ h0,
                       const float* __restrict__ c0, const int* __restrict__ enc_mask,
                       unsigned short* __restrict__ embs, unsigned short* __restrict__ xh,
                       float* __restrict__ c, float* __restrict__ mask_bias) {
  const long N_EMB = (long)B_ * TDEC * E_;   // 1,048,576
  const long N_ADD = (long)B_ * A_;          // 4,096
  const long N_H   = (long)B_ * H_;          // 32,768
  const long N_M   = (long)B_ * TENC;        // 4,096
  const long total = N_EMB + N_ADD + N_H + N_H + N_M;
  long i = (long)blockIdx.x * blockDim.x + threadIdx.x;
  long stride = (long)gridDim.x * blockDim.x;
  for (; i < total; i += stride) {
    long j = i;
    if (j < N_EMB) {
      long b = j / (TDEC * E_), r = j % (TDEC * E_);
      long t = r / E_, e = r % E_;
      int tok = (t == 0) ? 0 : input_var[b * TDEC + (t - 1)];
      embs[j] = f32_to_bf16(embed[(long)tok * E_ + e]);
      continue;
    }
    j -= N_EMB;
    if (j < N_ADD) {
      long b = j / A_, a = j % A_;
      xh[b * XH_W + E_ + a] = f32_to_bf16(add_var[j]);
      continue;
    }
    j -= N_ADD;
    if (j < N_H) {                            // h0 -> xh bf16
      long b = j / H_, k = j % H_;
      xh[b * XH_W + RNN_IN + k] = f32_to_bf16(h0[j]);
      continue;
    }
    j -= N_H;
    if (j < N_H) { c[j] = c0[j]; continue; }  // c0 -> c
    j -= N_H;
    mask_bias[j] = (enc_mask[j] > 0) ? 0.0f : -3.4028234663852886e+38f;
  }
}

// ---------------------------------------------------------------------------
// Host-side orchestration
// ---------------------------------------------------------------------------
extern "C" void kernel_launch(void* const* d_in, const int* in_sizes, int n_in,
                              void* d_out, int out_size, void* d_ws, size_t ws_size,
                              hipStream_t stream) {
  const int*   input_var = (const int*)  d_in[0];
  const float* add_var   = (const float*)d_in[1];
  const float* h0        = (const float*)d_in[2];
  const float* c0        = (const float*)d_in[3];
  const float* enc       = (const float*)d_in[4];
  const int*   enc_mask  = (const int*)  d_in[5];
  const float* embed     = (const float*)d_in[6];
  const float* Wa        = (const float*)d_in[7];
  const float* Wih       = (const float*)d_in[8];
  const float* Whh       = (const float*)d_in[9];
  const float* bih       = (const float*)d_in[10];
  const float* bhh       = (const float*)d_in[11];
  const float* Wout      = (const float*)d_in[12];
  const float* bout      = (const float*)d_in[13];
  float* out = (float*)d_out;                // (B, TDEC, V)

  // ---- workspace layout (256-byte aligned slices) ----
  char* ws = (char*)d_ws;
  size_t off = 0;
  auto carve = [&](size_t bytes) {
    char* p = ws + off;
    off += (bytes + 255) & ~size_t(255);
    return p;
  };
  unsigned short* Wout_b = (unsigned short*)carve((size_t)V_ * H_ * 2);    // 64 MB
  unsigned short* Wg_b   = (unsigned short*)carve((size_t)G4H * XH_W * 2); // 21 MB
  unsigned short* WaT_b  = (unsigned short*)carve((size_t)H_ * H_ * 2);    //  2 MB
  unsigned short* embs_b = (unsigned short*)carve((size_t)B_ * TDEC * E_ * 2);
  unsigned short* xh_b   = (unsigned short*)carve((size_t)B_ * XH_W * 2);
  float* q_ws    = (float*)carve((size_t)B_ * H_ * 4);
  float* gates   = (float*)carve((size_t)B_ * G4H * 4);
  float* c_state = (float*)carve((size_t)B_ * H_ * 4);
  float* mbias   = (float*)carve((size_t)B_ * TENC * 4);
  (void)ws_size; (void)n_in; (void)in_sizes; (void)out_size;

  // ---- one-time conversions (deterministic, re-run every call) ----
  k_cvt<<<1024, 256, 0, stream>>>(Wout, Wout_b, (long)V_ * H_);
  k_pack_wg<<<1024, 256, 0, stream>>>(Wih, Whh, Wg_b);
  k_transpose_wa<<<512, 256, 0, stream>>>(Wa, WaT_b);
  k_init<<<1024, 256, 0, stream>>>(input_var, embed, add_var, h0, c0, enc_mask,
                                   embs_b, xh_b, c_state, mbias);

  const unsigned short* h_b = xh_b + RNN_IN;   // bf16 hidden state, lda = XH_W

  // ---- 64 sequential decode steps ----
  for (int t = 0; t < TDEC; ++t) {
    // q = h @ Wa            (32x1024) = (32x1024)*(1024x1024)
    k_gemm_bf16<<<H_ / 256, 256, 0, stream>>>(h_b, XH_W, WaT_b, H_,
                                              nullptr, q_ws, (long)H_, H_);
    // attention + ctx -> xh[:, 640:1664], embedding -> xh[:, 0:512]
    k_attn<<<B_, TENC, 0, stream>>>(q_ws, enc, mbias, embs_b, xh_b, t);
    // gates = [x|h] @ [Wih|Whh]^T   (32x4096), K = 2688
    k_gemm_bf16<<<G4H / 256, 256, 0, stream>>>(xh_b, XH_W, Wg_b, XH_W,
                                               nullptr, gates, (long)G4H, G4H);
    // LSTM cell -> c, h2 (bf16 into xh)
    k_cell<<<(B_ * H_ + 255) / 256, 256, 0, stream>>>(gates, bih, bhh, c_state, xh_b);
    // logits = h2 @ Wout^T + bout  -> out[:, t, :]
    k_gemm_bf16<<<V_ / 256, 256, 0, stream>>>(h_b, XH_W, Wout_b, H_,
                                              bout, out + (long)t * V_,
                                              (long)TDEC * V_, V_);
  }
}
